// GINE_59554016526994
// MI455X (gfx1250) — matrix operand order repaired and checked by reading, not compiled, not run on hardware
//
#include <hip/hip_runtime.h>
#include <hip/hip_bf16.h>

#define C 128
#define ECH 16
#define LAYERS 4

typedef __attribute__((ext_vector_type(8)))  float  v8f;
typedef __attribute__((ext_vector_type(16))) __bf16 v16bf;

struct B32x2 { uint4 a, b; };

static __device__ __forceinline__ v16bf frag_bf16(uint4 a, uint4 b) {
    B32x2 u; u.a = a; u.b = b;
    return __builtin_bit_cast(v16bf, u);
}

// round-to-nearest-even f32 -> bf16, pack two into a dword (lo = first)
static __device__ __forceinline__ unsigned pack2_bf16(float x, float y) {
    unsigned ux = __float_as_uint(x);
    unsigned uy = __float_as_uint(y);
    ux = (ux + 0x7FFFu + ((ux >> 16) & 1u)) >> 16;
    uy = (uy + 0x7FFFu + ((uy >> 16) & 1u)) & 0xFFFF0000u;
    return uy | ux;
}
static __device__ __forceinline__ float bf16lo(unsigned p) { return __uint_as_float(p << 16); }
static __device__ __forceinline__ float bf16hi(unsigned p) { return __uint_as_float(p & 0xFFFF0000u); }
static __device__ __forceinline__ float silu_f(float v)    { return v / (1.0f + __expf(-v)); }

// ---------------------------------------------------------------------------
// ea = silu(edge_attr @ We + be), stored bf16-packed (fits MI455X 192MB L2)
// one thread per (edge, channel-pair)
// ---------------------------------------------------------------------------
__global__ void k_edge_embed(const float* __restrict__ eattr,
                             const float* __restrict__ We,
                             const float* __restrict__ be,
                             unsigned* __restrict__ ea_out, int E) {
    long gid = (long)blockIdx.x * blockDim.x + threadIdx.x;
    long total = (long)E * (C / 2);
    if (gid >= total) return;
    int e  = (int)(gid >> 6);
    int cp = (int)(gid & 63);
    int c0 = cp * 2;
    const float* row = eattr + (long)e * ECH;
    float a0 = be[c0], a1 = be[c0 + 1];
#pragma unroll
    for (int k = 0; k < ECH; ++k) {
        float a = row[k];
        a0 += a * We[k * C + c0];
        a1 += a * We[k * C + c0 + 1];
    }
    ea_out[gid] = pack2_bf16(silu_f(a0), silu_f(a1));
}

__global__ void k_zero4(float4* __restrict__ p, long n4) {
    long i = (long)blockIdx.x * blockDim.x + threadIdx.x;
    if (i < n4) p[i] = make_float4(0.f, 0.f, 0.f, 0.f);
}

// ---------------------------------------------------------------------------
// scatter: agg[dst] += relu(x[src] + ea)  (fp32 atomics into L2-resident agg)
// 4 threads per edge, 32 channels each
// ---------------------------------------------------------------------------
__global__ void k_scatter(const float* __restrict__ x,
                          const unsigned* __restrict__ ea,
                          const int* __restrict__ src,
                          const int* __restrict__ dst,
                          float* __restrict__ agg, int E) {
    long gid = (long)blockIdx.x * blockDim.x + threadIdx.x;
    long total = (long)E * 4;
    if (gid >= total) return;
    int e  = (int)(gid >> 2);
    int c0 = (int)(gid & 3) * 32;
    int s = src[e], d = dst[e];
    const float*    xr = x  + (long)s * C + c0;
    const unsigned* er = ea + (long)e * (C / 2) + (c0 >> 1);
    float*          ar = agg + (long)d * C + c0;
#pragma unroll
    for (int i = 0; i < 16; ++i) {
        unsigned p = er[i];
        float m0 = xr[2 * i]     + bf16lo(p);
        float m1 = xr[2 * i + 1] + bf16hi(p);
        if (m0 > 0.f)
            __hip_atomic_fetch_add(ar + 2 * i,     m0, __ATOMIC_RELAXED, __HIP_MEMORY_SCOPE_AGENT);
        if (m1 > 0.f)
            __hip_atomic_fetch_add(ar + 2 * i + 1, m1, __ATOMIC_RELAXED, __HIP_MEMORY_SCOPE_AGENT);
    }
}

// ---------------------------------------------------------------------------
// WMMA tile GEMM: acc[nt] += A(16x128 bf16, LDS [r][kpair]) x B(128x128, LDS [n][kpair])
// per ISA 16-bit A 16x32 / B 32x16 wave32 layouts, K split into 4 steps of 32
// ---------------------------------------------------------------------------
static __device__ __forceinline__ void gemm_tile(const unsigned* myA, const unsigned* sW,
                                                 int l15, int hf, v8f acc[8]) {
#pragma unroll
    for (int ks = 0; ks < 4; ++ks) {
        const unsigned* ap = myA + l15 * 64 + ks * 16 + hf * 4;
        v16bf afrag = frag_bf16(*(const uint4*)ap, *(const uint4*)(ap + 8));
#pragma unroll
        for (int nt = 0; nt < 8; ++nt) {
            const unsigned* bp = sW + (nt * 16 + l15) * 64 + ks * 16 + hf * 4;
            v16bf bfrag = frag_bf16(*(const uint4*)bp, *(const uint4*)(bp + 8));
            acc[nt] = __builtin_amdgcn_wmma_f32_16x16x32_bf16(
                false, afrag, false, bfrag, (short)0, acc[nt], false, false);
        }
    }
}

// ---------------------------------------------------------------------------
// fused layer: h = x+agg; t1 = silu(h@W1+b1); h2 = t1@W2+b2; LN(x+h2)
// 128 threads = 4 waves, each wave owns a 16-row tile (64 rows per block)
// ---------------------------------------------------------------------------
__global__ __launch_bounds__(128) void k_layer(
    const float* __restrict__ x_in, const float* __restrict__ agg,
    const float* __restrict__ W1l, const float* __restrict__ b1l,
    const float* __restrict__ W2l, const float* __restrict__ b2l,
    const float* __restrict__ gml, const float* __restrict__ btl,
    float* __restrict__ x_out, int N) {
    __shared__ unsigned sW[C * (C / 2)];        // 32KB: weights bf16, [n][kpair]
    __shared__ unsigned sA[4][16 * (C / 2)];    // 16KB: per-wave A staging [r][kpair]

    const int tid  = threadIdx.x;
    const int wave = tid >> 5, lane = tid & 31;
    const int l15  = lane & 15, hf = lane >> 4;
    const long tile0 = (long)blockIdx.x * 64 + wave * 16;
    unsigned* myA = sA[wave];

    // stage h = x + agg as packed bf16
#pragma unroll
    for (int i = 0; i < 32; ++i) {
        int p = lane + 32 * i;
        int r = p >> 6, j = p & 63;
        long row = tile0 + r; if (row >= N) row = N - 1;
        const float2 xv = *(const float2*)(x_in + row * C + 2 * j);
        const float2 av = *(const float2*)(agg  + row * C + 2 * j);
        myA[p] = pack2_bf16(xv.x + av.x, xv.y + av.y);
    }
    // stage W1 transposed: sW[n][j] = (W1[2j][n], W1[2j+1][n])
#pragma unroll
    for (int i = 0; i < 64; ++i) {
        int p = tid + 128 * i;
        int n = p >> 6, j = p & 63;
        sW[p] = pack2_bf16(W1l[(2 * j) * C + n], W1l[(2 * j + 1) * C + n]);
    }
    __syncthreads();

    v8f acc[8];
#pragma unroll
    for (int nt = 0; nt < 8; ++nt) {
        v8f z = {0.f, 0.f, 0.f, 0.f, 0.f, 0.f, 0.f, 0.f};
        acc[nt] = z;
    }
    gemm_tile(myA, sW, l15, hf, acc);
    __syncthreads();   // everyone done reading sW before overwrite

    // bias + SiLU, restage t1 into myA as A-layout bf16 (pair via lane xor-1)
#pragma unroll
    for (int nt = 0; nt < 8; ++nt) {
        float bias = b1l[nt * 16 + l15];
#pragma unroll
        for (int r8 = 0; r8 < 8; ++r8) {
            float v = silu_f(acc[nt][r8] + bias);
            float w = __shfl_xor(v, 1);
            if ((lane & 1) == 0) {
                int m = r8 + hf * 8;
                int n = nt * 16 + l15;
                myA[m * 64 + (n >> 1)] = pack2_bf16(v, w);
            }
        }
    }
    // stage W2
#pragma unroll
    for (int i = 0; i < 64; ++i) {
        int p = tid + 128 * i;
        int n = p >> 6, j = p & 63;
        sW[p] = pack2_bf16(W2l[(2 * j) * C + n], W2l[(2 * j + 1) * C + n]);
    }
    __syncthreads();

#pragma unroll
    for (int nt = 0; nt < 8; ++nt) {
        v8f z = {0.f, 0.f, 0.f, 0.f, 0.f, 0.f, 0.f, 0.f};
        acc[nt] = z;
    }
    gemm_tile(myA, sW, l15, hf, acc);

    // epilogue: + b2 + residual x, then LayerNorm per row (16 lanes per half)
#pragma unroll
    for (int nt = 0; nt < 8; ++nt) {
        float bias = b2l[nt * 16 + l15];
#pragma unroll
        for (int r8 = 0; r8 < 8; ++r8) {
            long row = tile0 + r8 + hf * 8;
            long rc  = row < N ? row : (N - 1);
            acc[nt][r8] += bias + x_in[rc * C + nt * 16 + l15];
        }
    }
#pragma unroll
    for (int r8 = 0; r8 < 8; ++r8) {
        float s = 0.f, s2 = 0.f;
#pragma unroll
        for (int nt = 0; nt < 8; ++nt) { float v = acc[nt][r8]; s += v; s2 += v * v; }
#pragma unroll
        for (int msk = 1; msk <= 8; msk <<= 1) {
            s  += __shfl_xor(s,  msk);
            s2 += __shfl_xor(s2, msk);
        }
        float mu  = s  * (1.0f / C);
        float var = s2 * (1.0f / C) - mu * mu;
        float rs  = rsqrtf(var + 1e-5f);
        long row = tile0 + r8 + hf * 8;
        if (row < N) {
#pragma unroll
            for (int nt = 0; nt < 8; ++nt) {
                int n = nt * 16 + l15;
                x_out[row * C + n] = (acc[nt][r8] - mu) * rs * gml[n] + btl[n];
            }
        }
    }
}

// ---------------------------------------------------------------------------
extern "C" void kernel_launch(void* const* d_in, const int* in_sizes, int n_in,
                              void* d_out, int out_size, void* d_ws, size_t ws_size,
                              hipStream_t stream) {
    const float* x      = (const float*)d_in[0];
    const float* eattr  = (const float*)d_in[1];
    const float* We     = (const float*)d_in[2];
    const float* be     = (const float*)d_in[3];
    const float* W1     = (const float*)d_in[4];   // [L][C][C]
    const float* b1     = (const float*)d_in[5];
    const float* W2     = (const float*)d_in[6];
    const float* b2     = (const float*)d_in[7];
    const float* gma    = (const float*)d_in[8];
    const float* bta    = (const float*)d_in[9];
    const int*   eidx   = (const int*)d_in[10];    // [2][E]

    const int N = in_sizes[0] / C;
    const int E = in_sizes[1] / ECH;
    const int* src = eidx;
    const int* dst = eidx + E;
    float* out = (float*)d_out;

    char* ws = (char*)d_ws;
    size_t off = 0;
    unsigned* ea = (unsigned*)(ws + off); off += (size_t)E * (C / 2) * sizeof(unsigned); // bf16 x2
    float* agg   = (float*)(ws + off);    off += (size_t)N * C * sizeof(float);
    float* xA    = (float*)(ws + off);    off += (size_t)N * C * sizeof(float);
    float* xB    = (float*)(ws + off);    off += (size_t)N * C * sizeof(float);

    // 1. edge embedding (shared across layers), bf16 so it stays L2-resident
    {
        long t = (long)E * (C / 2);
        k_edge_embed<<<dim3((unsigned)((t + 255) / 256)), dim3(256), 0, stream>>>(
            eattr, We, be, ea, E);
    }

    const float* xin = x;
    float* louts[LAYERS] = {xA, xB, xA, out};
    for (int l = 0; l < LAYERS; ++l) {
        long n4 = (long)N * (C / 4);
        k_zero4<<<dim3((unsigned)((n4 + 255) / 256)), dim3(256), 0, stream>>>((float4*)agg, n4);

        long ts = (long)E * 4;
        k_scatter<<<dim3((unsigned)((ts + 255) / 256)), dim3(256), 0, stream>>>(
            xin, ea, src, dst, agg, E);

        k_layer<<<dim3((unsigned)((N + 63) / 64)), dim3(128), 0, stream>>>(
            xin, agg,
            W1 + (size_t)l * C * C, b1 + (size_t)l * C,
            W2 + (size_t)l * C * C, b2 + (size_t)l * C,
            gma + (size_t)l * C, bta + (size_t)l * C,
            louts[l], N);
        xin = louts[l];
    }
}
